// OrthoLinear_27565100106050
// MI455X (gfx1250) — matrix operand order repaired and checked by reading, not compiled, not run on hardware
//
#include <hip/hip_runtime.h>
#include <hip/hip_fp16.h>

typedef _Float16 v16h __attribute__((ext_vector_type(16)));
typedef _Float16 v8h  __attribute__((ext_vector_type(8)));
typedef float    v8f  __attribute__((ext_vector_type(8)));
typedef float    v4f  __attribute__((ext_vector_type(4)));
typedef int      v4i  __attribute__((ext_vector_type(4)));

#define IN_F   1024
#define OUT_F  1024
#define TOKENS 32768
#define BK     32
#define NK     (IN_F / BK)
#define LDSW   48   // halves per LDS row: 32 data + 16 pad (96B rows, 32B aligned)

#if defined(__gfx1250__) && __has_builtin(__builtin_amdgcn_global_load_async_to_lds_b128)
#define HAVE_ASYNC_LDS 1
typedef __attribute__((address_space(1))) v4i g_v4i;   // global b128 chunk
typedef __attribute__((address_space(3))) v4i l_v4i;   // LDS b128 chunk
#else
#define HAVE_ASYNC_LDS 0
#endif

__device__ __forceinline__ void async_wait0() {
#if defined(__gfx1250__) && __has_builtin(__builtin_amdgcn_s_wait_asynccnt)
    __builtin_amdgcn_s_wait_asynccnt(0);
#elif HAVE_ASYNC_LDS
    asm volatile("s_wait_asynccnt 0" ::: "memory");
#endif
}

// ---------------------------------------------------------------------------
// Kernel 1: copy base weight (fp16) into workspace, 16B vector copies.
// ---------------------------------------------------------------------------
__global__ __launch_bounds__(256)
void prep_copy(const uint4* __restrict__ src, uint4* __restrict__ dst) {
    int t = blockIdx.x * 256 + threadIdx.x;
    dst[t] = src[t];
}

// ---------------------------------------------------------------------------
// Kernel 2: fold sparse ortho term: W[idx] = base[idx] + alpha * val.
// Indices are unique, so plain stores (WAW vs kernel 1 ordered by stream).
// ---------------------------------------------------------------------------
__global__ __launch_bounds__(256)
void prep_scatter(const _Float16* __restrict__ base,
                  const _Float16* __restrict__ vals,
                  const int* __restrict__ idx,
                  const float* __restrict__ alpha,
                  _Float16* __restrict__ wc, int nnz) {
    int t = blockIdx.x * 256 + threadIdx.x;
    if (t < nnz) {
        int i = idx[t];
        wc[i] = (_Float16)((float)base[i] + alpha[0] * (float)vals[t]);
    }
}

// ---------------------------------------------------------------------------
// Kernel 3: Y[m,n] = sum_k X[m,k] * Wc[n,k]   (M=32768, N=1024, K=1024)
// 128x128 workgroup tile, 8 waves, each wave 64x32 via 4x2 WMMA tiles.
// Double-buffered LDS; W tile staged with async LDS DMA when available.
// ---------------------------------------------------------------------------
__global__ __launch_bounds__(256)
void ortho_gemm(const float* __restrict__ X, const _Float16* __restrict__ W,
                float* __restrict__ Y) {
    __shared__ _Float16 Xs[2][128 * LDSW];
    __shared__ _Float16 Ws[2][128 * LDSW];

    const int tid   = threadIdx.x;
    const int lane  = tid & 31;
    const int wave  = tid >> 5;
    const int waveM = wave & 1;     // 2 slabs of 64 rows
    const int waveN = wave >> 1;    // 4 slabs of 32 cols
    const int l16   = lane & 15;
    const int lhalf = lane >> 4;    // 0 or 1

    const int blockN0 = blockIdx.x * 128;
    const int blockM0 = blockIdx.y * 128;

    // staging roles: 256 threads cover 128 rows x (2 x 16-wide) k segments
    const int srow = tid >> 1;      // 0..127
    const int sseg = tid & 1;       // which 16-wide half of the BK=32 slice

    const float*    gxr = X + (size_t)(blockM0 + srow) * IN_F + sseg * 16;
    const _Float16* gwr = W + (size_t)(blockN0 + srow) * IN_F + sseg * 16;

    v8f acc[4][2];
    for (int i = 0; i < 4; ++i)
        for (int j = 0; j < 2; ++j)
            acc[i][j] = (v8f){0.f, 0.f, 0.f, 0.f, 0.f, 0.f, 0.f, 0.f};

    auto stage = [&](int buf, int k0) {
        // ---- X tile: fp32 -> fp16 through registers ----
        const float* gx = gxr + k0;
        v4f f0 = *(const v4f*)(gx + 0);
        v4f f1 = *(const v4f*)(gx + 4);
        v4f f2 = *(const v4f*)(gx + 8);
        v4f f3 = *(const v4f*)(gx + 12);
        v8h h0, h1;
        for (int i = 0; i < 4; ++i) {
            h0[i]     = (_Float16)f0[i];
            h0[4 + i] = (_Float16)f1[i];
            h1[i]     = (_Float16)f2[i];
            h1[4 + i] = (_Float16)f3[i];
        }
        _Float16* xd = &Xs[buf][srow * LDSW + sseg * 16];
        *(v8h*)(xd + 0) = h0;
        *(v8h*)(xd + 8) = h1;

        // ---- W tile: straight fp16 copy ----
        const _Float16* gw = gwr + k0;
        _Float16*       wd = &Ws[buf][srow * LDSW + sseg * 16];
#if HAVE_ASYNC_LDS
        __builtin_amdgcn_global_load_async_to_lds_b128(
            (g_v4i*)(gw + 0), (l_v4i*)(wd + 0), 0, 0);
        __builtin_amdgcn_global_load_async_to_lds_b128(
            (g_v4i*)(gw + 8), (l_v4i*)(wd + 8), 0, 0);
#else
        v8h w0 = *(const v8h*)(gw + 0);
        v8h w1 = *(const v8h*)(gw + 8);
        *(v8h*)(wd + 0) = w0;
        *(v8h*)(wd + 8) = w1;
#endif
    };

    auto compute = [&](int buf) {
        v16h afrag[4];
        const int akoff = lhalf * 16;              // 16 contiguous K per lane
        for (int mt = 0; mt < 4; ++mt) {
            int row = waveM * 64 + mt * 16 + l16;
            v8h lo = *(const v8h*)(&Xs[buf][row * LDSW + akoff]);
            v8h hi = *(const v8h*)(&Xs[buf][row * LDSW + akoff + 8]);
            afrag[mt] = __builtin_shufflevector(lo, hi,
                0, 1, 2, 3, 4, 5, 6, 7, 8, 9, 10, 11, 12, 13, 14, 15);
        }
        v16h bfrag[2];
        const int bkoff = lhalf * 16;              // 16 contiguous K per lane
        for (int nt = 0; nt < 2; ++nt) {
            int col = waveN * 32 + nt * 16 + l16;
            bfrag[nt] = *(const v16h*)(&Ws[buf][col * LDSW + bkoff]);
        }
        for (int mt = 0; mt < 4; ++mt)
            for (int nt = 0; nt < 2; ++nt)
                acc[mt][nt] = __builtin_amdgcn_wmma_f32_16x16x32_f16(
                    false, afrag[mt], false, bfrag[nt],
                    (short)0, acc[mt][nt], false, false);
    };

    // prologue: fill buffer 0
    stage(0, 0);
    async_wait0();
    __syncthreads();

#pragma unroll 2
    for (int ks = 0; ks < NK; ++ks) {
        const int cur = ks & 1;
        if (ks + 1 < NK)
            stage(cur ^ 1, (ks + 1) * BK);   // issue next tile's loads early
        compute(cur);                        // 8 WMMAs overlap the loads
        async_wait0();                       // async W stores to nxt done
        __syncthreads();                     // reads of cur done, nxt visible
    }

    // ---- store: C layout -> row = r + 8*lhalf, col = l16 within each tile ----
    for (int mt = 0; mt < 4; ++mt) {
        for (int nt = 0; nt < 2; ++nt) {
            const int col     = blockN0 + waveN * 32 + nt * 16 + l16;
            const int rowBase = blockM0 + waveM * 64 + mt * 16 + lhalf * 8;
            for (int r = 0; r < 8; ++r) {
                Y[(size_t)(rowBase + r) * OUT_F + col] = acc[mt][nt][r];
            }
        }
    }
}

// ---------------------------------------------------------------------------
extern "C" void kernel_launch(void* const* d_in, const int* in_sizes, int n_in,
                              void* d_out, int out_size, void* d_ws, size_t ws_size,
                              hipStream_t stream) {
    (void)in_sizes; (void)n_in; (void)out_size; (void)ws_size;

    const float*    x     = (const float*)d_in[0];
    const _Float16* bw    = (const _Float16*)d_in[1];
    const _Float16* ov    = (const _Float16*)d_in[2];
    const int*      oi    = (const int*)d_in[3];
    const float*    alpha = (const float*)d_in[4];
    float*          out   = (float*)d_out;
    _Float16*       wc    = (_Float16*)d_ws;   // 1024*1024 fp16 = 2 MB

    // 1) W_comb = W_base   (2MB as 131072 uint4 -> 512 blocks x 256 threads)
    prep_copy<<<512, 256, 0, stream>>>((const uint4*)bw, (uint4*)wc);
    // 2) W_comb[idx] = base[idx] + alpha * val
    prep_scatter<<<64, 256, 0, stream>>>(bw, ov, oi, alpha, wc, 16384);
    // 3) GEMM. N-blocks fastest so X stripes are L2-reused across N.
    dim3 grid(OUT_F / 128, TOKENS / 128);
    ortho_gemm<<<grid, 256, 0, stream>>>(x, wc, out);
}
